// CustomLSTMForecast_91259465106005
// MI455X (gfx1250) — matrix operand order repaired and checked by reading, not compiled, not run on hardware
//
#include <hip/hip_runtime.h>

// -------- problem constants (B, T, I, H) = (512, 256, 256, 512) --------
#define B_ 512
#define T_ 256
#define I_ 256
#define H_ 512
#define G_ (4 * H_)        // 2048 gate rows
#define K_ (H_ + I_)       // 768 contraction length: [h | x_t]

typedef __attribute__((ext_vector_type(16))) __bf16 v16bf;
typedef __attribute__((ext_vector_type(8)))  __bf16 v8bf;
typedef __attribute__((ext_vector_type(8)))  float  v8f;

struct bf16x4 { __bf16 a, b, c, d; };   // 8-byte packed store

// ---------------- fast activations (straight-line, no libm branches) --------
__device__ __forceinline__ float fast_sigmoid(float x) {
    return 1.0f / (1.0f + __expf(-x));
}
__device__ __forceinline__ float fast_tanh(float x) {
    x = fminf(15.0f, fmaxf(-15.0f, x));
    float e = __expf(2.0f * x);
    return (e - 1.0f) / (e + 1.0f);
}

// ---------------- f32 -> bf16 conversion, 4 elements/thread ----------------
__global__ void cvt_f32_to_bf16_x4(const float4* __restrict__ src,
                                   bf16x4* __restrict__ dst, int n4) {
    int i = blockIdx.x * blockDim.x + threadIdx.x;
    if (i < n4) {
        float4 v = src[i];
        bf16x4 o;
        o.a = (__bf16)v.x; o.b = (__bf16)v.y;
        o.c = (__bf16)v.z; o.d = (__bf16)v.w;
        dst[i] = o;
    }
}

// ---------------- zero h0 (bf16) and c0 (f32) ----------------
__global__ void init_state(__bf16* __restrict__ h0, float* __restrict__ c, int n) {
    int i = blockIdx.x * blockDim.x + threadIdx.x;
    if (i < n) { h0[i] = (__bf16)0.0f; c[i] = 0.0f; }
}

// ---------------- one LSTM timestep ----------------
// grid = (B/16, H/32), block = 128 threads = 4 waves (one gate group per wave).
// Each wave computes a 16(batch) x 32(cell) f32 gate tile: one A fragment is
// reused across two v_wmma_f32_16x16x32_bf16 issues per k-step (K = 768).
// Gate tiles are exchanged through LDS, then the block applies the
// sigmoid/tanh cell update for its (batch-tile, cell-tile).
__global__ __launch_bounds__(128)
void lstm_step(const __bf16* __restrict__ xb,    // (B, T, I) bf16
               const __bf16* __restrict__ Wb16,  // (G, K) bf16, rows = [Wh | Wx]
               const float*  __restrict__ bias,  // (G,)  f32
               const __bf16* __restrict__ h_in,  // (B, H) bf16  (state t)
               __bf16*       __restrict__ h_out, // (B, H) bf16  (state t+1)
               float*        __restrict__ c,     // (B, H) f32   (in/out)
               int t)
{
    __shared__ float gl[4][16][32];   // [gate][batch-row][cell-col]  (8 KB)

    const int lane = threadIdx.x & 31;
    const int wave = threadIdx.x >> 5;        // gate group 0..3 (f, i, o, c_hat)
    const int b0   = blockIdx.x * 16;         // batch tile origin
    const int n0   = blockIdx.y * 32;         // cell tile origin (within H)

    const int half = lane >> 4;               // 0: lanes 0-15, 1: lanes 16-31
    const int l15  = lane & 15;

    // A fragment: row M = b0 + l15 of [h | x_t]
    const int arow = b0 + l15;
    // B fragments: lane -> weight row (column of B); two 16-col tiles
    const __bf16* wrow0 = Wb16 + (size_t)(wave * H_ + n0 + l15) * K_;
    const __bf16* wrow1 = wrow0 + (size_t)16 * K_;

    v8f acc0 = {};
    v8f acc1 = {};

    #pragma unroll 4
    for (int kk = 0; kk < K_; kk += 32) {
        // ---- A: 16x32 bf16 fragment (lanes 0-15 hold K+0..15,
        // lanes 16-31 hold K+16..31 of the same row) ----
        const __bf16* abase;
        if (kk < H_) {
            abase = h_in + (size_t)arow * H_;                     // h part
        } else {
            abase = xb + ((size_t)arow * T_ + t) * I_ - H_;       // x_t part
        }
        const int ca0 = kk + (half ? 16 : 0);
        const int ca1 = kk + (half ? 24 : 8);
        v8bf a_lo = *(const v8bf*)(abase + ca0);
        v8bf a_hi = *(const v8bf*)(abase + ca1);
        v16bf a;
        #pragma unroll
        for (int e = 0; e < 8; ++e) { a[e] = a_lo[e]; a[e + 8] = a_hi[e]; }

        // ---- B: two 32x16 bf16 fragments (lane = output column; lanes 0-15
        // hold K+0..15, lanes 16-31 hold K+16..31 -> 32 contiguous B/lane) ----
        const int cb = kk + (half ? 16 : 0);
        v16bf bf0 = *(const v16bf*)(wrow0 + cb);
        v16bf bf1 = *(const v16bf*)(wrow1 + cb);

        // prefetch next weight chunks (emits global_prefetch_b8)
        __builtin_prefetch((const void*)(wrow0 + cb + 64), 0, 1);
        __builtin_prefetch((const void*)(wrow1 + cb + 64), 0, 1);

        acc0 = __builtin_amdgcn_wmma_f32_16x16x32_bf16(
                   false, a, false, bf0, (short)0, acc0, false, false);
        acc1 = __builtin_amdgcn_wmma_f32_16x16x32_bf16(
                   false, a, false, bf1, (short)0, acc1, false, false);
    }

    // C/D layout: VGPR r, lanes 0-15 -> (M=r, N=lane); lanes 16-31 -> (M=r+8).
    #pragma unroll
    for (int r = 0; r < 8; ++r) {
        gl[wave][r + half * 8][l15]      = acc0[r];
        gl[wave][r + half * 8][16 + l15] = acc1[r];
    }
    __syncthreads();

    // ---- elementwise cell update: 512 elements over 128 threads ----
    #pragma unroll
    for (int e = 0; e < 4; ++e) {
        int idx = threadIdx.x + e * 128;     // 0..511
        int bi  = idx >> 5;                  // batch row in tile
        int ni  = idx & 31;                  // cell col in tile
        int kc  = n0 + ni;                   // cell index within H

        float f  = fast_sigmoid(gl[0][bi][ni] + bias[kc]);
        float ig = fast_sigmoid(gl[1][bi][ni] + bias[H_ + kc]);
        float o  = fast_sigmoid(gl[2][bi][ni] + bias[2 * H_ + kc]);
        float ch = fast_tanh  (gl[3][bi][ni] + bias[3 * H_ + kc]);

        size_t off = (size_t)(b0 + bi) * H_ + kc;
        float cn = f * c[off] + ig * ch;     // c stays f32 across all steps
        c[off] = cn;
        h_out[off] = (__bf16)(o * fast_tanh(cn));
    }
}

// ---------------- final fc head: out[b] = h_T[b,:] . fc_w + fc_b ----------------
__global__ __launch_bounds__(128)
void fc_head(const __bf16* __restrict__ hT, const float* __restrict__ fw,
             const float* __restrict__ fb, float* __restrict__ out)
{
    __shared__ float red[128];
    int b = blockIdx.x;
    float s = 0.0f;
    for (int k = threadIdx.x; k < H_; k += 128)
        s += (float)hT[(size_t)b * H_ + k] * fw[k];
    red[threadIdx.x] = s;
    __syncthreads();
    for (int st = 64; st > 0; st >>= 1) {
        if (threadIdx.x < st) red[threadIdx.x] += red[threadIdx.x + st];
        __syncthreads();
    }
    if (threadIdx.x == 0) out[b] = red[0] + fb[0];
}

// ---------------- host-side launcher ----------------
extern "C" void kernel_launch(void* const* d_in, const int* in_sizes, int n_in,
                              void* d_out, int out_size, void* d_ws, size_t ws_size,
                              hipStream_t stream)
{
    const float* x    = (const float*)d_in[0];   // (B, T, I)
    const float* W_w  = (const float*)d_in[1];   // (4H, H+I) = (G, K)
    const float* W_b  = (const float*)d_in[2];   // (G,)
    const float* fc_w = (const float*)d_in[3];   // (1, H)
    const float* fc_b = (const float*)d_in[4];   // (1,)
    float* out = (float*)d_out;                  // (B, 1)

    // workspace layout (all offsets 256B-aligned)
    char* ws = (char*)d_ws;
    size_t off = 0;
    __bf16* xb   = (__bf16*)(ws + off); off += (size_t)B_ * T_ * I_ * 2;  // 64 MB
    __bf16* Wb16 = (__bf16*)(ws + off); off += (size_t)G_ * K_ * 2;       // 3 MB
    __bf16* hbuf = (__bf16*)(ws + off); off += (size_t)2 * B_ * H_ * 2;   // 1 MB (double buffer)
    float*  cbuf = (float*)(ws + off);  off += (size_t)B_ * H_ * 4;       // 1 MB
    (void)ws_size; (void)in_sizes; (void)n_in; (void)out_size;

    // one-time per call: bf16 copies of x and W, zero h0/c0
    {
        int n4 = (B_ * T_ * I_) / 4;
        cvt_f32_to_bf16_x4<<<(n4 + 255) / 256, 256, 0, stream>>>(
            (const float4*)x, (bf16x4*)xb, n4);
    }
    {
        int n4 = (G_ * K_) / 4;
        cvt_f32_to_bf16_x4<<<(n4 + 255) / 256, 256, 0, stream>>>(
            (const float4*)W_w, (bf16x4*)Wb16, n4);
    }
    {
        int n = B_ * H_;
        init_state<<<(n + 255) / 256, 256, 0, stream>>>(hbuf, cbuf, n);
    }

    // sequential scan: one launch per timestep (launch = device-wide barrier)
    for (int t = 0; t < T_; ++t) {
        const __bf16* hin  = hbuf + (size_t)(t & 1) * B_ * H_;
        __bf16*       hout = hbuf + (size_t)((t + 1) & 1) * B_ * H_;
        lstm_step<<<dim3(B_ / 16, H_ / 32), 128, 0, stream>>>(
            xb, Wb16, W_b, hin, hout, cbuf, t);
    }

    // h_T lives in buffer (T & 1) == 0
    fc_head<<<B_, 128, 0, stream>>>(hbuf + (size_t)(T_ & 1) * B_ * H_,
                                    fc_w, fc_b, out);
}
